// conv_layer_65000035058096
// MI455X (gfx1250) — matrix operand order repaired and checked by reading, not compiled, not run on hardware
//
#include <hip/hip_runtime.h>
#include <hip/hip_bf16.h>
#include <stdint.h>

// ---------------------------------------------------------------------------
// Spherical 1-ring conv (ico6): out[b,v,:] = W @ concat_{n=0..6} x[b, idx[v,n], :] + bias
// bf16 WMMA 16x16x32; K=448 = 14 k-tiles, N=64 = 4 n-tiles.
// Main kernel: one wave = TWO 16-row tiles (32 rows) x 64 cols, exactly
//   10240 full pairs (no bounds checks at all). 32-bit element offsets
//   against uniform SGPR bases (saddr + voffset addressing).
// Tail kernel: last 4 rows, scalar fp32 (negligible).
// W prepacked into per-lane B fragments (ws), staged once per block in LDS.
// K loop kept rolled so B streams from LDS (no VGPR blowup / spills).
// ---------------------------------------------------------------------------

typedef __attribute__((ext_vector_type(16))) __bf16 v16bf;
typedef __attribute__((ext_vector_type(8)))  float  v8f;
typedef __attribute__((ext_vector_type(4)))  float  v4f;

#define NV      163842
#define CIN     64
#define COUT    64
#define KTILES  14          // 448 / 32
#define NTILES  4           // 64 / 16
#define NFRAG   (KTILES * NTILES * 32)          // fragments * lanes
#define WPACK_ELEMS (NFRAG * 16)                // bf16 elements = 28672
#define TOTAL_ROWS (2 * NV)                     // 327684
#define FULL_PAIRS (TOTAL_ROWS / 32)            // 10240 (32-row pairs)
#define TAIL_START (FULL_PAIRS * 32)            // 327680
#define TAIL_ROWS  (TOTAL_ROWS - TAIL_START)    // 4

// --- prepack W (64x448 f32, [n][k]) into B fragments -----------------------
// Fragment f = (kt*NTILES + nt), lane l, element e holds B[k][n] = W[n][k]
// with k = kt*32 + (l>=16 ? 8 : 0) + (e < 8 ? e : e + 8),  n = nt*16 + (l & 15).
__global__ __launch_bounds__(256) void prep_w_kernel(
    const float* __restrict__ W, __bf16* __restrict__ wpack)
{
    int i = blockIdx.x * blockDim.x + threadIdx.x;
    if (i >= WPACK_ELEMS) return;
    int e    = i & 15;
    int lane = (i >> 4) & 31;
    int nt   = (i >> 9) & 3;
    int kt   = i >> 11;                 // 0..13
    int k = kt * 32 + ((lane >= 16) ? 8 : 0) + ((e < 8) ? e : e + 8);
    int n = nt * 16 + (lane & 15);
    wpack[i] = (__bf16)W[n * 448 + k];
}

// Build one A fragment for this lane from x + 32-bit element offset.
// Lanes 0-15: K {0..7, 16..23}; lanes 16-31: K {8..15, 24..31} (pre-offset).
__device__ __forceinline__ v16bf load_a_frag(const float* __restrict__ x,
                                             uint32_t eoff)
{
    v4f a0 = *(const v4f*)(x + eoff);
    v4f a1 = *(const v4f*)(x + eoff + 4);
    v4f a2 = *(const v4f*)(x + eoff + 16);
    v4f a3 = *(const v4f*)(x + eoff + 20);
    v16bf af;
#pragma unroll
    for (int e = 0; e < 4; ++e) {
        af[e]      = (__bf16)a0[e];
        af[4 + e]  = (__bf16)a1[e];
        af[8 + e]  = (__bf16)a2[e];
        af[12 + e] = (__bf16)a3[e];
    }
    return af;
}

// --- main fused gather + GEMM (full pairs only, no bounds checks) ----------
__global__ __launch_bounds__(256) void conv7_wmma_kernel(
    const float*   __restrict__ x,
    const int32_t* __restrict__ nbr32,   // int64 neigh_orders viewed as i32 pairs
    const float*   __restrict__ bias,
    const v16bf*   __restrict__ wpack,
    float*         __restrict__ out)
{
    __shared__ v16bf lds_b[NFRAG];       // 57,344 bytes

    // Stage all B fragments into LDS once per block.
    for (int i = threadIdx.x; i < NFRAG; i += blockDim.x)
        lds_b[i] = wpack[i];
    __syncthreads();

    const int  lane = threadIdx.x & 31;
    const int  wave = threadIdx.x >> 5;
    const int  m    = lane & 15;
    const bool hi   = lane >= 16;
    const uint32_t koff = hi ? 8u : 0u;

    // bias for the column this lane stores (N = nt*16 + m)
    float bv[NTILES];
#pragma unroll
    for (int nt = 0; nt < NTILES; ++nt) bv[nt] = bias[nt * 16 + m];

    for (int pair = blockIdx.x * 8 + wave; pair < FULL_PAIRS;
         pair += gridDim.x * 8) {
        const uint32_t g0 = (uint32_t)pair * 32u;    // first row of tile 0
        // Per-lane gathered row offsets (elements into x) for both tiles.
        uint32_t rowoff[2][7];
#pragma unroll
        for (int t = 0; t < 2; ++t) {
            const uint32_t g = g0 + (uint32_t)(t * 16) + (uint32_t)m;
            const uint32_t b = (g >= (uint32_t)NV) ? 1u : 0u;
            const uint32_t v = g - b * (uint32_t)NV;
#pragma unroll
            for (int n = 0; n < 7; ++n) {
                uint32_t idx = (uint32_t)nbr32[((size_t)v * 7 + n) * 2];
                rowoff[t][n] = (b * (uint32_t)NV + idx) * (uint32_t)CIN;
            }
        }
#pragma unroll
        for (int t = 0; t < 2; ++t)
#pragma unroll
            for (int n = 0; n < 7; ++n)
                __builtin_prefetch(x + rowoff[t][n], 0, 3); // global_prefetch_b8

        v8f acc[2][NTILES] = {};

#pragma unroll 1        // keep rolled: B streams from LDS, no VGPR blowup
        for (int kt = 0; kt < KTILES; ++kt) {
            const uint32_t off = (uint32_t)((kt & 1) * 32) + koff;
            const int n7 = kt >> 1;
            v16bf af0 = load_a_frag(x, rowoff[0][n7] + off);
            v16bf af1 = load_a_frag(x, rowoff[1][n7] + off);
#pragma unroll
            for (int nt = 0; nt < NTILES; ++nt) {
                v16bf bf = lds_b[(kt * NTILES + nt) * 32 + lane];
                acc[0][nt] = __builtin_amdgcn_wmma_f32_16x16x32_bf16(
                    false, af0, false, bf, (short)0, acc[0][nt], false, false);
                acc[1][nt] = __builtin_amdgcn_wmma_f32_16x16x32_bf16(
                    false, af1, false, bf, (short)0, acc[1][nt], false, false);
            }
        }

        // Store: D layout — VGPR r: lanes 0-15 row r, lanes 16-31 row r+8;
        // col = lane & 15. Single 32-bit base offset, immediate offsets only.
        float* __restrict__ outp =
            out + (size_t)((g0 + (hi ? 8u : 0u)) * (uint32_t)COUT + (uint32_t)m);
#pragma unroll
        for (int t = 0; t < 2; ++t)
#pragma unroll
            for (int nt = 0; nt < NTILES; ++nt)
#pragma unroll
                for (int r = 0; r < 8; ++r)
                    outp[(t * 16 + r) * COUT + nt * 16] =
                        acc[t][nt][r] + bv[nt];
    }
}

// --- tail: last TAIL_ROWS rows, scalar fp32 --------------------------------
__global__ __launch_bounds__(256) void conv7_tail_kernel(
    const float*   __restrict__ x,
    const int32_t* __restrict__ nbr32,
    const float*   __restrict__ W,
    const float*   __restrict__ bias,
    float*         __restrict__ out)
{
    int i = blockIdx.x * blockDim.x + threadIdx.x;   // over TAIL_ROWS*COUT
    if (i >= TAIL_ROWS * COUT) return;
    int r = i / COUT, o = i % COUT;
    int g = TAIL_START + r;
    int b = (g >= NV) ? 1 : 0;
    int v = g - b * NV;
    float s = bias[o];
#pragma unroll
    for (int n = 0; n < 7; ++n) {
        int idx = nbr32[((size_t)v * 7 + n) * 2];
        const float* xr = x + ((size_t)b * NV + idx) * CIN;
        const float* wr = W + o * 448 + n * CIN;
        float acc = 0.f;
        for (int c = 0; c < CIN; ++c) acc += xr[c] * wr[c];
        s += acc;
    }
    out[(size_t)g * COUT + o] = s;
}

extern "C" void kernel_launch(void* const* d_in, const int* in_sizes, int n_in,
                              void* d_out, int out_size, void* d_ws, size_t ws_size,
                              hipStream_t stream) {
    const float*   x    = (const float*)d_in[0];
    const int32_t* nbr  = (const int32_t*)d_in[1];   // int64 viewed as i32 pairs (LE)
    const float*   W    = (const float*)d_in[2];
    const float*   bias = (const float*)d_in[3];
    float*         out  = (float*)d_out;
    __bf16*        wpk  = (__bf16*)d_ws;             // 57,344 bytes used

    (void)in_sizes; (void)n_in; (void)out_size; (void)ws_size;

    prep_w_kernel<<<(WPACK_ELEMS + 255) / 256, 256, 0, stream>>>(W, wpk);

    const int blocks = FULL_PAIRS / 8;               // 1280, 1 pair per wave
    conv7_wmma_kernel<<<blocks, 256, 0, stream>>>(
        x, nbr, bias, (const v16bf*)wpk, out);

    conv7_tail_kernel<<<1, 256, 0, stream>>>(x, nbr, W, bias, out);
}
